// MultiHeadedAttention_6502580486520
// MI455X (gfx1250) — matrix operand order repaired and checked by reading, not compiled, not run on hardware
//
#include <hip/hip_runtime.h>

// MI455X / gfx1250: wave32, WMMA 16x16x32 bf16 (f32 accum).
// MHA: B=2 S=2048 E=1024 H=16 L=64, causal softmax.

#define BATCH 2
#define SEQ   2048
#define EMB   1024
#define NH    16
#define HD    64

typedef __attribute__((ext_vector_type(16))) __bf16 v16bf;
typedef __attribute__((ext_vector_type(2)))  __bf16 v2bf;
typedef __attribute__((ext_vector_type(8)))  float  v8f;

// ---- WMMA helpers -----------------------------------------------------------

__device__ __forceinline__ v8f wmma_bf16(v16bf a, v16bf b, v8f c) {
  // D = A(16x32) * B(32x16) + C(16x16 f32)
  return __builtin_amdgcn_wmma_f32_16x16x32_bf16(false, a, false, b, (short)0, c,
                                                 false, false);
}

// 16-bit A-matrix 16x32 layout (contiguous, matching B):
//   lanes 0-15  (half=0): VGPR g -> K = 2g,2g+1
//   lanes 16-31 (half=1): VGPR g -> K = 16+2g,16+2g+1
__device__ __forceinline__ int a_k0(int g, int half) {
  return half * 16 + 2 * g;
}

// Load an A-fragment for row (lane&15) from a row-major bf16 row pointer.
__device__ __forceinline__ v16bf load_a_frag(const __bf16* rowp, int half) {
  v16bf a;
#pragma unroll
  for (int g = 0; g < 8; ++g) {
    v2bf t = *(const v2bf*)(rowp + a_k0(g, half));
    a[2 * g]     = t[0];
    a[2 * g + 1] = t[1];
  }
  return a;
}

// B-matrix 32x16: lane n=lane&15 holds column n; lanes 0-15 K=0..15,
// lanes 16-31 K=16..31, contiguous per lane -> single 32B load.

// Row reductions across the 16 lanes holding one D-fragment row.
__device__ __forceinline__ float rmax16(float v) {
#pragma unroll
  for (int m = 1; m <= 8; m <<= 1) v = fmaxf(v, __shfl_xor(v, m, 16));
  return v;
}
__device__ __forceinline__ float rsum16(float v) {
#pragma unroll
  for (int m = 1; m <= 8; m <<= 1) v += __shfl_xor(v, m, 16);
  return v;
}

// ---- elementwise f32 -> bf16 ------------------------------------------------

__global__ void cvt_f32_bf16(const float* __restrict__ s, __bf16* __restrict__ d,
                             int n) {
  int idx = (blockIdx.x * blockDim.x + threadIdx.x) * 4;
  if (idx + 3 < n) {
    float4 v = *(const float4*)(s + idx);
    d[idx]     = (__bf16)v.x;
    d[idx + 1] = (__bf16)v.y;
    d[idx + 2] = (__bf16)v.z;
    d[idx + 3] = (__bf16)v.w;
  }
}

// ---- fused QKV projection GEMM ---------------------------------------------
// C[m=(b,s)][n=(qkv,h,l)] = sum_e xb[m][e] * Wqkv[n][e]  (+bias at store)
// grid: (256 m-tiles, 12) x 128 threads; each wave: 1 m-tile x 4 n-tiles.

__global__ void __launch_bounds__(128) qkv_gemm_kernel(
    const __bf16* __restrict__ xb, const __bf16* __restrict__ wb,
    const float* __restrict__ bq, const float* __restrict__ bk,
    const float* __restrict__ bv, __bf16* __restrict__ Qb,
    __bf16* __restrict__ Kb, __bf16* __restrict__ Vtb) {
  const int lane = threadIdx.x & 31;
  const int wid  = threadIdx.x >> 5;
  const int nn   = lane & 15;
  const int half = lane >> 4;
  const int mtile  = blockIdx.x;               // 16 rows of [B*S]
  const int ngroup = blockIdx.y * 4 + wid;     // 64 cols of [3*H*L]

  v8f acc[4] = {};
  const __bf16* arow = xb + (size_t)(mtile * 16 + nn) * EMB;
  for (int kk = 0; kk < EMB; kk += 32) {
    v16bf a = load_a_frag(arow + kk, half);
#pragma unroll
    for (int f = 0; f < 4; ++f) {
      int n = ngroup * 64 + f * 16 + nn;
      v16bf b = *(const v16bf*)(wb + (size_t)n * EMB + kk + half * 16);
      acc[f] = wmma_bf16(a, b, acc[f]);
    }
  }

#pragma unroll
  for (int f = 0; f < 4; ++f) {
    int n   = ngroup * 64 + f * 16 + nn;
    int qkv = n >> 10;
    int rem = n & 1023;                 // h*64 + l
    int hh = rem >> 6, ll = rem & 63;
    float bias = (qkv == 0 ? bq : (qkv == 1 ? bk : bv))[rem];
#pragma unroll
    for (int r = 0; r < 8; ++r) {
      int rowg = mtile * 16 + r + 8 * half;   // b*SEQ + s
      int bb = rowg >> 11, ss = rowg & 2047;
      __bf16 v = (__bf16)(acc[f][r] + bias);
      if (qkv == 0)
        Qb[((size_t)(bb * NH + hh) * SEQ + ss) * HD + ll] = v;       // [B,H,S,L]
      else if (qkv == 1)
        Kb[((size_t)(bb * NH + hh) * SEQ + ss) * HD + ll] = v;       // [B,H,S,L]
      else
        Vtb[((size_t)(bb * NH + hh) * HD + ll) * SEQ + ss] = v;      // [B,H,L,S]
    }
  }
}

// ---- flash attention --------------------------------------------------------
// grid: (B*H=32, S/64=32) x 128 threads; each wave: 16 query rows, 32 keys/iter.

__global__ void __launch_bounds__(128) flash_attn_kernel(
    const __bf16* __restrict__ Qb, const __bf16* __restrict__ Kb,
    const __bf16* __restrict__ Vtb, __bf16* __restrict__ zb) {
  __shared__ __bf16 ldsP[4][16][40];   // per-wave 16x32 P tile (+pad)

  const int lane = threadIdx.x & 31;
  const int wid  = threadIdx.x >> 5;
  const int nn   = lane & 15;
  const int half = lane >> 4;
  const int bh = blockIdx.x;
  const int b = bh >> 4, h = bh & 15;
  const int qs = blockIdx.y * 64 + wid * 16;

  const __bf16* Qp = Qb + (size_t)bh * SEQ * HD;
  const __bf16* Kp = Kb + (size_t)bh * SEQ * HD;
  const __bf16* Vp = Vtb + (size_t)bh * HD * SEQ;

  const __bf16* qrow = Qp + (size_t)(qs + nn) * HD;
  v16bf qa0 = load_a_frag(qrow, half);        // l = 0..31
  v16bf qa1 = load_a_frag(qrow + 32, half);   // l = 32..63

  v8f o[4] = {};
  float mrow[8], lrow[8];
#pragma unroll
  for (int r = 0; r < 8; ++r) { mrow[r] = -3.0e38f; lrow[r] = 0.0f; }

  const float scl = 0.125f;  // 1/sqrt(64)

  for (int t0 = 0; t0 <= qs + 15; t0 += 32) {        // causal key range
    const __bf16* k0p = Kp + (size_t)(t0 + nn) * HD + half * 16;
    const __bf16* k1p = Kp + (size_t)(t0 + 16 + nn) * HD + half * 16;
    v16bf kb00 = *(const v16bf*)(k0p);
    v16bf kb01 = *(const v16bf*)(k0p + 32);
    v16bf kb10 = *(const v16bf*)(k1p);
    v16bf kb11 = *(const v16bf*)(k1p + 32);

    v8f s0 = {}, s1 = {};
    s0 = wmma_bf16(qa0, kb00, s0);
    s0 = wmma_bf16(qa1, kb01, s0);
    s1 = wmma_bf16(qa0, kb10, s1);
    s1 = wmma_bf16(qa1, kb11, s1);

#pragma unroll
    for (int r = 0; r < 8; ++r) {
      int rowq = qs + r + 8 * half;
      float a = s0[r] * scl;
      if (t0 + nn > rowq) a = -3.0e38f;                // causal mask
      float c = s1[r] * scl;
      if (t0 + 16 + nn > rowq) c = -3.0e38f;
      float mx   = rmax16(fmaxf(a, c));
      float mnew = fmaxf(mrow[r], mx);
      float alpha = __expf(mrow[r] - mnew);
      mrow[r] = mnew;
      float p0 = __expf(a - mnew);
      float p1 = __expf(c - mnew);
      lrow[r] = lrow[r] * alpha + rsum16(p0 + p1);
#pragma unroll
      for (int f = 0; f < 4; ++f) o[f][r] *= alpha;
      ldsP[wid][r + 8 * half][nn]      = (__bf16)p0;  // D-layout -> LDS
      ldsP[wid][r + 8 * half][16 + nn] = (__bf16)p1;
    }
    // Wave-private LDS transpose; DS ops are in-order per wave, just stop
    // compiler reordering and drain DScnt before A-fragment reads.
    asm volatile("s_wait_dscnt 0x0" ::: "memory");
    v16bf pa = load_a_frag(&ldsP[wid][nn][0], half);   // P as 16x32 A-frag
    asm volatile("" ::: "memory");

#pragma unroll
    for (int f = 0; f < 4; ++f) {
      v16bf vb = *(const v16bf*)(Vp + (size_t)(f * 16 + nn) * SEQ + t0 + half * 16);
      o[f] = wmma_bf16(pa, vb, o[f]);
    }
  }

#pragma unroll
  for (int r = 0; r < 8; ++r) {
    float inv = 1.0f / lrow[r];
    int srow = qs + r + 8 * half;
    size_t base = ((size_t)(b * SEQ + srow)) * (NH * HD) + h * HD;
#pragma unroll
    for (int f = 0; f < 4; ++f)
      zb[base + f * 16 + nn] = (__bf16)(o[f][r] * inv);  // z: [B,S,H*L]
  }
}

// ---- output projection GEMM -------------------------------------------------
// out[m=(b,s)][e] = sum_k z[m][k] * Wo[e][k] + bo[e]
// grid: (256, 4) x 128 threads; each wave: 1 m-tile x 4 n-tiles.

__global__ void __launch_bounds__(128) out_gemm_kernel(
    const __bf16* __restrict__ zb, const __bf16* __restrict__ wob,
    const float* __restrict__ bo, float* __restrict__ out) {
  const int lane = threadIdx.x & 31;
  const int wid  = threadIdx.x >> 5;
  const int nn   = lane & 15;
  const int half = lane >> 4;
  const int mtile  = blockIdx.x;
  const int ngroup = blockIdx.y * 4 + wid;

  v8f acc[4] = {};
  const __bf16* arow = zb + (size_t)(mtile * 16 + nn) * (NH * HD);
  for (int kk = 0; kk < NH * HD; kk += 32) {
    v16bf a = load_a_frag(arow + kk, half);
#pragma unroll
    for (int f = 0; f < 4; ++f) {
      int e = ngroup * 64 + f * 16 + nn;
      v16bf bfr = *(const v16bf*)(wob + (size_t)e * (NH * HD) + kk + half * 16);
      acc[f] = wmma_bf16(a, bfr, acc[f]);
    }
  }

#pragma unroll
  for (int f = 0; f < 4; ++f) {
    int e = ngroup * 64 + f * 16 + nn;
    float bias = bo[e];
#pragma unroll
    for (int r = 0; r < 8; ++r) {
      int rowg = mtile * 16 + r + 8 * half;
      out[(size_t)rowg * EMB + e] = acc[f][r] + bias;
    }
  }
}

// ---- launch -----------------------------------------------------------------

extern "C" void kernel_launch(void* const* d_in, const int* in_sizes, int n_in,
                              void* d_out, int out_size, void* d_ws, size_t ws_size,
                              hipStream_t stream) {
  const float* x  = (const float*)d_in[0];
  // d_in[1] = mask (causal, baked into kernel)
  const float* Wq = (const float*)d_in[2];
  const float* bq = (const float*)d_in[3];
  const float* Wk = (const float*)d_in[4];
  const float* bk = (const float*)d_in[5];
  const float* Wv = (const float*)d_in[6];
  const float* bv = (const float*)d_in[7];
  const float* Wo = (const float*)d_in[8];
  const float* bo = (const float*)d_in[9];
  float* out = (float*)d_out;

  // Workspace layout (bf16 elements); total ~48 MB.
  __bf16* ws  = (__bf16*)d_ws;
  __bf16* xb  = ws;                  // [B*S, E]        4,194,304
  __bf16* wqb = xb  + 4194304;       // [H*L, E]        1,048,576
  __bf16* wkb = wqb + 1048576;       //                 (wq/wk/wv contiguous ->
  __bf16* wvb = wkb + 1048576;       //                  one [3072, E] matrix)
  __bf16* wob = wvb + 1048576;       // [E, H*L]        1,048,576
  __bf16* Qb  = wob + 1048576;       // [B,H,S,L]       4,194,304
  __bf16* Kb  = Qb  + 4194304;       // [B,H,S,L]
  __bf16* Vtb = Kb  + 4194304;       // [B,H,L,S]
  __bf16* zb  = Vtb + 4194304;       // [B,S,H*L]

  cvt_f32_bf16<<<4096, 256, 0, stream>>>(x,  xb,  4194304);
  cvt_f32_bf16<<<1024, 256, 0, stream>>>(Wq, wqb, 1048576);
  cvt_f32_bf16<<<1024, 256, 0, stream>>>(Wk, wkb, 1048576);
  cvt_f32_bf16<<<1024, 256, 0, stream>>>(Wv, wvb, 1048576);
  cvt_f32_bf16<<<1024, 256, 0, stream>>>(Wo, wob, 1048576);

  qkv_gemm_kernel<<<dim3(256, 12), 128, 0, stream>>>(xb, wqb, bq, bk, bv,
                                                     Qb, Kb, Vtb);
  flash_attn_kernel<<<dim3(32, 32), 128, 0, stream>>>(Qb, Kb, Vtb, zb);
  out_gemm_kernel<<<dim3(256, 4), 128, 0, stream>>>(zb, wob, bo, out);
}
